// MultiHeadAttention_46763603918869
// MI455X (gfx1250) — matrix operand order repaired and checked by reading, not compiled, not run on hardware
//
#include <hip/hip_runtime.h>
#include <hip/hip_bf16.h>
#include <math.h>

// MHA forward, B=2, S=2048, D=1024, H=16, HD=64 on gfx1250 (wave32, WMMA bf16).
// d_out = [ctx: B*S*D fp32][scores: B*H*S*S fp32]

typedef __attribute__((ext_vector_type(16))) __bf16 v16bf;
typedef __attribute__((ext_vector_type(8)))  float  v8f;

constexpr int Bc = 2, Sc = 2048, Dc = 1024, Hc = 16, HDc = 64;
constexpr size_t NX = (size_t)Bc * Sc * Dc; // 4,194,304 (q/k/v elements)
constexpr size_t NW = (size_t)Dc * Dc;      // 1,048,576 (weight elements)

// ---------------------------------------------------------------------------
// Kernel 0: one-shot fp32 -> bf16 conversion of activations and weights so
// the GEMM inner loops are pure bf16 b128 loads (no cvt chains, half bytes).
// grid.y in [0,6): {q,k,v,Wq,Wk,Wv}. Each thread converts 8 elements.
// ---------------------------------------------------------------------------
__global__ __launch_bounds__(256) void cvt_bf16_kernel(
    const float* __restrict__ q,  const float* __restrict__ k,
    const float* __restrict__ v,  const float* __restrict__ Wq,
    const float* __restrict__ Wk, const float* __restrict__ Wv,
    __bf16* __restrict__ Xb, __bf16* __restrict__ Wb)
{
    const int sel = blockIdx.y;
    const float* src;
    __bf16* dst;
    size_t n;
    if (sel < 3) { // activations
        src = (sel == 0) ? q : (sel == 1) ? k : v;
        dst = Xb + (size_t)sel * NX;
        n = NX;
    } else {       // weights
        src = (sel == 3) ? Wq : (sel == 4) ? Wk : Wv;
        dst = Wb + (size_t)(sel - 3) * NW;
        n = NW;
    }
    const size_t base = ((size_t)blockIdx.x * blockDim.x + threadIdx.x) * 8;
    if (base >= n) return;
    #pragma unroll
    for (int i = 0; i < 8; ++i) dst[base + i] = (__bf16)src[base + i];
}

// ---------------------------------------------------------------------------
// Kernel 1: QKV projections. Y = X @ W^T + b with bf16 operands, f32 acc.
// One wave computes a 32(M) x 64(N) tile: 2 A-subtiles x 4 B-subtiles =
// 8 WMMAs per 32-wide k-step (32 steps -> 256 WMMA/wave).
// Outputs: Qh/Kh bf16 [B,H,S,HD]; V transposed Vt bf16 [B,H,HD,S].
// ---------------------------------------------------------------------------
__global__ __launch_bounds__(256) void qkv_proj_kernel(
    const __bf16* __restrict__ Xb, const __bf16* __restrict__ Wb,
    const float* __restrict__ bq, const float* __restrict__ bk,
    const float* __restrict__ bv,
    __bf16* __restrict__ Qh, __bf16* __restrict__ Kh, __bf16* __restrict__ Vt)
{
    const int wave = (blockIdx.x * blockDim.x + threadIdx.x) >> 5;
    const int lane = threadIdx.x & 31;
    const int Mtot    = Bc * Sc;            // 4096
    const int tilesN4 = Dc / 64;            // 16
    const int tilesM2 = Mtot / 32;          // 128
    const int per_mat = tilesM2 * tilesN4;  // 2048 waves per matrix

    const int which = wave / per_mat;       // 0=Q 1=K 2=V (uniform per wave)
    const int rem   = wave % per_mat;
    const int mtile = rem / tilesN4;
    const int nt4   = rem % tilesN4;

    const __bf16* X    = Xb + (size_t)which * NX;
    const __bf16* W    = Wb + (size_t)which * NW;
    const float*  bias = (which == 0) ? bq : (which == 1) ? bk : bv;

    const int r  = lane & 15;   // A: row in tile; B: column; C/D: N
    const int kg = lane >> 4;   // K-half selector

    v8f c[2][4] = {};
    for (int kk = 0; kk < Dc; kk += 32) {
        // Two A subtiles (rows mtile*32 + mt*16 + r).
        v16bf a[2];
        #pragma unroll
        for (int mt = 0; mt < 2; ++mt) {
            const int mrow = mtile * 32 + mt * 16 + r;
            const __bf16* xr = X + (size_t)mrow * Dc + kk + kg * 16;
            #pragma unroll
            for (int i = 0; i < 8; ++i) a[mt][i]     = xr[i];      // K = 16kg+i
            #pragma unroll
            for (int i = 0; i < 8; ++i) a[mt][8 + i] = xr[8 + i]; // K = 16kg+8+i
        }
        #pragma unroll
        for (int j = 0; j < 4; ++j) {
            const int ncol = nt4 * 64 + j * 16 + r;   // W row = output column
            const __bf16* wr = W + (size_t)ncol * Dc + kk + kg * 16;
            v16bf bm;
            #pragma unroll
            for (int i = 0; i < 16; ++i) bm[i] = wr[i];
            #pragma unroll
            for (int mt = 0; mt < 2; ++mt)
                c[mt][j] = __builtin_amdgcn_wmma_f32_16x16x32_bf16(
                    false, a[mt], false, bm, (short)0, c[mt][j], false, false);
        }
    }

    // Epilogue: bias, split heads, store bf16 (V transposed).
    #pragma unroll
    for (int j = 0; j < 4; ++j) {
        const int ncol = nt4 * 64 + j * 16 + r;       // same N for all rows
        const float bval = bias[ncol];
        const int h = ncol >> 6, hd = ncol & 63;
        #pragma unroll
        for (int mt = 0; mt < 2; ++mt) {
            #pragma unroll
            for (int rr = 0; rr < 8; ++rr) {
                const int Mg = mtile * 32 + mt * 16 + rr + kg * 8;
                const int b = Mg >> 11, s = Mg & (Sc - 1);
                const float val = c[mt][j][rr] + bval;
                if (which == 0)
                    Qh[(((size_t)(b * Hc + h) * Sc) + s) * HDc + hd] = (__bf16)val;
                else if (which == 1)
                    Kh[(((size_t)(b * Hc + h) * Sc) + s) * HDc + hd] = (__bf16)val;
                else
                    Vt[(((size_t)(b * Hc + h) * HDc) + hd) * Sc + s] = (__bf16)val;
            }
        }
    }
}

// ---------------------------------------------------------------------------
// Kernel 2: scores = Qh @ Kh^T * (1/sqrt(HD)) + mask -> d_out (fp32).
// One wave: 32(M) x 64(N) tile of one (b,h); K-loop over HD=64 ->
// 2 steps x 8 WMMA = 16 WMMA/wave.
// ---------------------------------------------------------------------------
__global__ __launch_bounds__(256) void scores_kernel(
    const __bf16* __restrict__ Qh, const __bf16* __restrict__ Kh,
    const float* __restrict__ mask, float* __restrict__ scores)
{
    const int wave = (blockIdx.x * blockDim.x + threadIdx.x) >> 5;
    const int lane = threadIdx.x & 31;
    const int tilesM2 = Sc / 32;  // 64
    const int tilesN4 = Sc / 64;  // 32

    const int bh    = wave / (tilesM2 * tilesN4);
    const int rem   = wave % (tilesM2 * tilesN4);
    const int mtile = rem / tilesN4;
    const int nt4   = rem % tilesN4;
    const int b = bh >> 4;

    const int r = lane & 15, kg = lane >> 4;
    const __bf16* Qb = Qh + (size_t)bh * Sc * HDc;
    const __bf16* Kb = Kh + (size_t)bh * Sc * HDc;

    v8f c[2][4] = {};
    #pragma unroll
    for (int kk = 0; kk < HDc; kk += 32) {
        v16bf a[2];
        #pragma unroll
        for (int mt = 0; mt < 2; ++mt) {
            const int mrow = mtile * 32 + mt * 16 + r;
            const __bf16* qr = Qb + (size_t)mrow * HDc + kk + kg * 16;
            #pragma unroll
            for (int i = 0; i < 8; ++i) a[mt][i]     = qr[i];
            #pragma unroll
            for (int i = 0; i < 8; ++i) a[mt][8 + i] = qr[8 + i];
        }
        #pragma unroll
        for (int j = 0; j < 4; ++j) {
            const int ncol = nt4 * 64 + j * 16 + r;
            const __bf16* kr = Kb + (size_t)ncol * HDc + kk + kg * 16;
            v16bf bm;
            #pragma unroll
            for (int i = 0; i < 16; ++i) bm[i] = kr[i];
            #pragma unroll
            for (int mt = 0; mt < 2; ++mt)
                c[mt][j] = __builtin_amdgcn_wmma_f32_16x16x32_bf16(
                    false, a[mt], false, bm, (short)0, c[mt][j], false, false);
        }
    }

    const float scale = 0.125f; // 1/sqrt(64)
    float* so = scores + (size_t)bh * Sc * Sc;
    #pragma unroll
    for (int j = 0; j < 4; ++j) {
        const int ncol = nt4 * 64 + j * 16 + r;
        const float mval = mask[b * Sc + ncol]; // [B,1,1,S] broadcast over rows
        #pragma unroll
        for (int mt = 0; mt < 2; ++mt) {
            #pragma unroll
            for (int rr = 0; rr < 8; ++rr) {
                const int Mg = mtile * 32 + mt * 16 + rr + kg * 8;
                so[(size_t)Mg * Sc + ncol] = c[mt][j][rr] * scale + mval;
            }
        }
    }
}

// ---------------------------------------------------------------------------
// Kernel 3: per-row softmax stats (max, sum of exp). One wave32 per row.
// ---------------------------------------------------------------------------
__global__ __launch_bounds__(256) void rowstats_kernel(
    const float* __restrict__ scores,
    float* __restrict__ rowmax, float* __restrict__ rowsum)
{
    const int row  = (blockIdx.x * blockDim.x + threadIdx.x) >> 5;
    const int lane = threadIdx.x & 31;
    const float* rp = scores + (size_t)row * Sc;

    float mx = -INFINITY;
    #pragma unroll 4
    for (int i = lane; i < Sc; i += 32) mx = fmaxf(mx, rp[i]);
    #pragma unroll
    for (int off = 16; off > 0; off >>= 1) mx = fmaxf(mx, __shfl_xor(mx, off, 32));

    float sum = 0.f;
    #pragma unroll 4
    for (int i = lane; i < Sc; i += 32) sum += __expf(rp[i] - mx);
    #pragma unroll
    for (int off = 16; off > 0; off >>= 1) sum += __shfl_xor(sum, off, 32);

    if (lane == 0) { rowmax[row] = mx; rowsum[row] = sum; }
}

// ---------------------------------------------------------------------------
// Kernel 4: ctx = softmax(scores) @ V. Probs recomputed on the fly from row
// stats (no 268 MB probs tensor). One wave: 16 rows x full HD=64, K-loop over
// all 2048 keys in steps of 32 (4 WMMA per step, 256 total per wave).
// ---------------------------------------------------------------------------
__global__ __launch_bounds__(256) void ctx_kernel(
    const float* __restrict__ scores, const __bf16* __restrict__ Vt,
    const float* __restrict__ rowmax, const float* __restrict__ rowsum,
    float* __restrict__ ctx)
{
    const int wave = (blockIdx.x * blockDim.x + threadIdx.x) >> 5;
    const int lane = threadIdx.x & 31;
    const int tilesM = Sc / 16; // 128
    const int bh    = wave / tilesM;
    const int mtile = wave % tilesM;
    const int b = bh >> 4, h = bh & 15;

    const int r = lane & 15, kg = lane >> 4;
    const int mrow = mtile * 16 + r;                 // A-operand row per lane
    const float rm  = rowmax[bh * Sc + mrow];
    const float inv = 1.f / rowsum[bh * Sc + mrow];
    const float* srow = scores + ((size_t)bh * Sc + mrow) * Sc;
    const __bf16* Vb  = Vt + (size_t)bh * HDc * Sc;

    v8f c[4] = {};
    for (int n0 = 0; n0 < Sc; n0 += 32) {
        v16bf a;
        #pragma unroll
        for (int i = 0; i < 8; ++i)
            a[i] = (__bf16)(__expf(srow[n0 + kg * 16 + i] - rm) * inv);
        #pragma unroll
        for (int i = 0; i < 8; ++i)
            a[8 + i] = (__bf16)(__expf(srow[n0 + 8 + kg * 16 + i] - rm) * inv);

        #pragma unroll
        for (int j = 0; j < 4; ++j) {
            const int d = j * 16 + r;                // output column (head dim)
            const __bf16* vr = Vb + (size_t)d * Sc + n0 + kg * 16; // contiguous K
            v16bf bm;
            #pragma unroll
            for (int i = 0; i < 16; ++i) bm[i] = vr[i];
            c[j] = __builtin_amdgcn_wmma_f32_16x16x32_bf16(
                false, a, false, bm, (short)0, c[j], false, false);
        }
    }

    // Store ctx fp32, merging heads back to [B,S,D].
    #pragma unroll
    for (int j = 0; j < 4; ++j) {
        const int d = j * 16 + r;
        #pragma unroll
        for (int rr = 0; rr < 8; ++rr) {
            const int s = mtile * 16 + rr + kg * 8;
            ctx[(((size_t)b * Sc + s) * Dc) + h * HDc + d] = c[j][rr];
        }
    }
}

// ---------------------------------------------------------------------------
extern "C" void kernel_launch(void* const* d_in, const int* in_sizes, int n_in,
                              void* d_out, int out_size, void* d_ws, size_t ws_size,
                              hipStream_t stream) {
    const float* q    = (const float*)d_in[0];
    const float* k    = (const float*)d_in[1];
    const float* v    = (const float*)d_in[2];
    const float* mask = (const float*)d_in[3];
    const float* Wq   = (const float*)d_in[4];
    const float* bq   = (const float*)d_in[5];
    const float* Wk   = (const float*)d_in[6];
    const float* bk   = (const float*)d_in[7];
    const float* Wv   = (const float*)d_in[8];
    const float* bv   = (const float*)d_in[9];

    float* ctx    = (float*)d_out;
    float* scores = (float*)d_out + (size_t)Bc * Sc * Dc; // ctx first, then scores

    // Workspace layout (bf16 unless noted):
    //   Xb[3*NX] | Wb[3*NW] | Qh[NX] | Kh[NX] | Vt[NX] | rowmax f32 | rowsum f32
    __bf16* Xb = (__bf16*)d_ws;
    __bf16* Wb = Xb + 3 * NX;
    __bf16* Qh = Wb + 3 * NW;
    __bf16* Kh = Qh + NX;
    __bf16* Vt = Kh + NX;
    float*  rowmax = (float*)(Vt + NX);          // B*H*S = 65536 floats
    float*  rowsum = rowmax + (size_t)Bc * Hc * Sc;

    // 0) fp32 -> bf16 conversion. x-blocks sized for the 4M activations
    //    (4M / (256*8) = 2048); weight slices bound-check out early.
    cvt_bf16_kernel<<<dim3(2048, 6), 256, 0, stream>>>(q, k, v, Wq, Wk, Wv, Xb, Wb);

    // 1) QKV projections: 3 * (4096/32) * (1024/64) = 6144 waves -> 768 blocks
    qkv_proj_kernel<<<768, 256, 0, stream>>>(Xb, Wb, bq, bk, bv, Qh, Kh, Vt);

    // 2) Scores: 32 * 64 * 32 = 65536 waves -> 8192 blocks
    scores_kernel<<<8192, 256, 0, stream>>>(Qh, Kh, mask, scores);

    // 3) Row stats: 65536 rows -> 8192 blocks
    rowstats_kernel<<<8192, 256, 0, stream>>>(scores, rowmax, rowsum);

    // 4) Context: 32 * 128 = 4096 waves -> 512 blocks
    ctx_kernel<<<512, 256, 0, stream>>>(scores, Vt, rowmax, rowsum, ctx);
}
